// TransformerClassifier_14362370638539
// MI455X (gfx1250) — matrix operand order repaired and checked by reading, not compiled, not run on hardware
//
#include <hip/hip_runtime.h>
#include <hip/hip_bf16.h>
#include <math.h>

// Problem constants (from the reference)
#define NN 100000   // nodes (divisible by 16 -> 6250 m-tiles, and by 8)
#define EE 1600000  // edges (divisible by 8)
#define DD 128      // hidden dim
#define HH 4        // heads
#define CC 32       // head dim
#define GG 4096     // graphs
#define LL 3        // stacked blocks (total layers = 1 + LL = 4)

typedef __attribute__((ext_vector_type(16))) __bf16 v16bf;
typedef __attribute__((ext_vector_type(8)))  float  v8f;
typedef __attribute__((ext_vector_type(4)))  float  f4;
typedef __attribute__((ext_vector_type(4)))  unsigned int u32x4;
typedef __attribute__((ext_vector_type(4)))  unsigned short u16x4;

__device__ __forceinline__ unsigned short f2bf(float f) {
  unsigned int u = __float_as_uint(f);
  u += 0x7FFFu + ((u >> 16) & 1u);  // round-to-nearest-even
  return (unsigned short)(u >> 16);
}

// float atomic-max via int/uint trick (init location to -inf)
__device__ __forceinline__ void atomicMaxF(float* addr, float v) {
  if (v >= 0.0f) atomicMax((int*)addr, __float_as_int(v));
  else           atomicMin((unsigned int*)addr, __float_as_uint(v));
}

// ---------------- utility kernels ----------------
__global__ void fill_f32(float* __restrict__ p, float v, long n) {
  long i = (long)blockIdx.x * blockDim.x + threadIdx.x;
  if (i < n) p[i] = v;
}

__global__ void f32_to_bf16_k(const float* __restrict__ x,
                              unsigned short* __restrict__ xb, long n) {
  long i = (long)blockIdx.x * blockDim.x + threadIdx.x;
  if (i < n) xb[i] = f2bf(x[i]);
}

// Pack Wq|Wk|Wv|Ws ([128 in x 128 out] each, row-major) into transposed bf16
// WcatT [512 out x 128 k] plus concatenated bias (512).
__global__ void pack_weights(const float* __restrict__ Wq, const float* __restrict__ Wk,
                             const float* __restrict__ Wv, const float* __restrict__ Ws,
                             const float* __restrict__ bq, const float* __restrict__ bk,
                             const float* __restrict__ bv, const float* __restrict__ bs,
                             unsigned short* __restrict__ wt, float* __restrict__ bias) {
  int t = blockIdx.x * blockDim.x + threadIdx.x;  // 512*128 threads
  int n = t >> 7;        // output column 0..511
  int k = t & 127;       // input k 0..127
  int s = n >> 7;        // section: 0=q,1=k,2=v,3=skip(Ws)
  int col = n & 127;
  const float* W = (s == 0) ? Wq : (s == 1) ? Wk : (s == 2) ? Wv : Ws;
  wt[n * DD + k] = f2bf(W[k * DD + col]);
  if (k == 0) {
    const float* b = (s == 0) ? bq : (s == 1) ? bk : (s == 2) ? bv : bs;
    bias[n] = b[col];
  }
}

// ---------------- WMMA GEMM: QKVS[N,512] = xb[N,128](bf16) @ WcatT^T + bias --
// One wave computes a 16(M) x 64(N) strip: 4 k-steps x 4 n-tiles = 16 WMMAs.
// A-frag layout (16-bit A 16x32): lane<16 holds M=lane, K in {k0..k0+7, k0+16..k0+23};
// lane>=16 same M, K shifted by 8 -> two contiguous 16B chunks per lane.
// B-frag layout (16-bit B 32x16): lane holds column N=lane&15, K same as A-frag (two 16B chunks)
// contiguous because weights are stored transposed [out][k].
__global__ void __launch_bounds__(256)
gemm_qkvs(const unsigned short* __restrict__ xb,
          const unsigned short* __restrict__ wt,
          const float* __restrict__ bias,
          float* __restrict__ out) {
  const int lane  = threadIdx.x & 31;
  const int wid   = blockIdx.x * 8 + (threadIdx.x >> 5);
  const int mtile = wid >> 3;      // 0..6249
  const int ngrp  = wid & 7;       // 0..7 -> 64-wide strip
  const long mbase = (long)mtile * 16;
  const int  nbase = ngrp * 64;
  const int mA = lane & 15;
  const int kb = (lane >> 4) * 8;
  const int nB = lane & 15;
  const int kh = (lane >> 4) * 8;
  const unsigned short* arow = xb + (mbase + mA) * DD;

  union AB { v16bf v; u32x4 q[2]; };
  v8f acc[4] = {v8f{0.f}, v8f{0.f}, v8f{0.f}, v8f{0.f}};

#pragma unroll
  for (int ks = 0; ks < 4; ++ks) {
    const int k0 = ks * 32;
    AB a;
    a.q[0] = *(const u32x4*)(arow + k0 + kb);
    a.q[1] = *(const u32x4*)(arow + k0 + 16 + kb);
#pragma unroll
    for (int nt = 0; nt < 4; ++nt) {
      const unsigned short* brow = wt + (long)(nbase + nt * 16 + nB) * DD + k0 + kh;
      AB b;
      b.q[0] = *(const u32x4*)(brow);
      b.q[1] = *(const u32x4*)(brow + 16);
      acc[nt] = __builtin_amdgcn_wmma_f32_16x16x32_bf16(
          false, a.v, false, b.v, (short)0, acc[nt], false, false);
    }
  }

  const int Moff = (lane >> 4) * 8;  // C/D: lanes>=16 hold rows M=8..15
#pragma unroll
  for (int nt = 0; nt < 4; ++nt) {
    const int col = nbase + nt * 16 + nB;
    const float bia = bias[col];
#pragma unroll
    for (int r = 0; r < 8; ++r) {
      out[(mbase + Moff + r) * 512 + col] = acc[nt][r] + bia;
    }
  }
}

// ---------------- edge pass A: alpha[e,h] = q[dst,h].k[src,h]/sqrt(C); seg-max
__global__ void __launch_bounds__(256)
edge_alpha(const int* __restrict__ src, const int* __restrict__ dst,
           const float* __restrict__ qkvs, float* __restrict__ alphab,
           float* __restrict__ mmax) {
  const int lane = threadIdx.x & 31;
  const long e = (long)blockIdx.x * 8 + (threadIdx.x >> 5);
  if (e >= EE) return;
  const int s = src[e], d = dst[e];
  const int idx = lane * 4;
  f4 q = *(const f4*)(qkvs + (long)d * 512 + idx);         // q at offset 0
  f4 k = *(const f4*)(qkvs + (long)s * 512 + 128 + idx);   // k at offset 128
  float p = q.x * k.x + q.y * k.y + q.z * k.z + q.w * k.w;
  p += __shfl_xor(p, 1, 32);
  p += __shfl_xor(p, 2, 32);
  p += __shfl_xor(p, 4, 32);        // 8-lane groups == one head (32 ch / 4 per lane)
  const int h = lane >> 3;
  if ((lane & 7) == 0) {
    float a = p * 0.17677669529663687f;  // 1/sqrt(32)
    alphab[e * 4 + h] = a;
    atomicMaxF(mmax + (long)d * 4 + h, a);
  }
}

// ---------------- edge pass B: e = exp(alpha - m[dst]); seg-sum denom --------
__global__ void edge_exp(const int* __restrict__ dst, float* __restrict__ alphab,
                         const float* __restrict__ mmax, float* __restrict__ denom) {
  long t = (long)blockIdx.x * blockDim.x + threadIdx.x;  // over E*H
  if (t >= (long)EE * 4) return;
  long e = t >> 2; int h = (int)(t & 3);
  int d = dst[e];
  float ex = __expf(alphab[t] - mmax[(long)d * 4 + h]);
  alphab[t] = ex;
  atomicAdd(denom + (long)d * 4 + h, ex);
}

// ---------------- edge pass C: out[dst] += a * v[src] ------------------------
__global__ void __launch_bounds__(256)
edge_msg(const int* __restrict__ src, const int* __restrict__ dst,
         const float* __restrict__ qkvs, const float* __restrict__ alphab,
         const float* __restrict__ denom, float* __restrict__ outagg) {
  const int lane = threadIdx.x & 31;
  const long e = (long)blockIdx.x * 8 + (threadIdx.x >> 5);
  if (e >= EE) return;
  const int s = src[e], d = dst[e];
  const int idx = lane * 4;
  const int h = lane >> 3;
  float a = alphab[e * 4 + h] / (denom[(long)d * 4 + h] + 1e-16f);
  f4 v = *(const f4*)(qkvs + (long)s * 512 + 256 + idx);   // v at offset 256
  float* o = outagg + (long)d * 128 + idx;
  atomicAdd(o + 0, a * v.x);
  atomicAdd(o + 1, a * v.y);
  atomicAdd(o + 2, a * v.z);
  atomicAdd(o + 3, a * v.w);
}

// ---------------- beta gate + ReLU + bf16 emit (wave per node) ---------------
__global__ void __launch_bounds__(256)
beta_blend(const float* __restrict__ outagg, const float* __restrict__ qkvs,
           const float* __restrict__ wbeta, float* __restrict__ h,
           unsigned short* __restrict__ xb) {
  const int lane = threadIdx.x & 31;
  const long n = (long)blockIdx.x * 8 + (threadIdx.x >> 5);
  if (n >= NN) return;
  const int idx = lane * 4;
  f4 o  = *(const f4*)(outagg + n * 128 + idx);
  f4 xr = *(const f4*)(qkvs + n * 512 + 384 + idx);        // skip-proj at 384
  float p = o.x * wbeta[idx]     + o.y * wbeta[idx + 1]
          + o.z * wbeta[idx + 2] + o.w * wbeta[idx + 3];
  p += xr.x * wbeta[128 + idx]     + xr.y * wbeta[128 + idx + 1]
     + xr.z * wbeta[128 + idx + 2] + xr.w * wbeta[128 + idx + 3];
  f4 dmx = o - xr;
  p += dmx.x * wbeta[256 + idx]     + dmx.y * wbeta[256 + idx + 1]
     + dmx.z * wbeta[256 + idx + 2] + dmx.w * wbeta[256 + idx + 3];
  p += __shfl_xor(p, 1, 32);
  p += __shfl_xor(p, 2, 32);
  p += __shfl_xor(p, 4, 32);
  p += __shfl_xor(p, 8, 32);
  p += __shfl_xor(p, 16, 32);      // all lanes now hold the full dot
  float beta = 1.0f / (1.0f + __expf(-p));
  f4 r;
  r.x = fmaxf(beta * xr.x + (1.0f - beta) * o.x, 0.0f);
  r.y = fmaxf(beta * xr.y + (1.0f - beta) * o.y, 0.0f);
  r.z = fmaxf(beta * xr.z + (1.0f - beta) * o.z, 0.0f);
  r.w = fmaxf(beta * xr.w + (1.0f - beta) * o.w, 0.0f);
  *(f4*)(h + n * 128 + idx) = r;
  u16x4 rb;
  rb.x = f2bf(r.x); rb.y = f2bf(r.y); rb.z = f2bf(r.z); rb.w = f2bf(r.w);
  *(u16x4*)(xb + n * 128 + idx) = rb;
}

// ---------------- global mean pool ------------------------------------------
__global__ void pool_accum(const int* __restrict__ batch, const float* __restrict__ h,
                           float* __restrict__ psum) {
  long t = (long)blockIdx.x * blockDim.x + threadIdx.x;  // N*128
  if (t >= (long)NN * 128) return;
  long n = t >> 7; int c = (int)(t & 127);
  atomicAdd(psum + (long)batch[n] * 128 + c, h[t]);
}

__global__ void cnt_accum(const int* __restrict__ batch, float* __restrict__ pcnt) {
  long n = (long)blockIdx.x * blockDim.x + threadIdx.x;
  if (n >= NN) return;
  atomicAdd(pcnt + batch[n], 1.0f);
}

// ---------------- MLP head (wave per graph) ----------------------------------
__global__ void __launch_bounds__(256)
mlp_head(const float* __restrict__ psum, const float* __restrict__ pcnt,
         const float* __restrict__ W1, const float* __restrict__ b1,
         const float* __restrict__ W2, const float* __restrict__ b2,
         const float* __restrict__ W3, const float* __restrict__ b3,
         float* __restrict__ out) {
  __shared__ float sg[8][128];
  __shared__ float l1[8][64];
  const int w = threadIdx.x >> 5;
  const int lane = threadIdx.x & 31;
  const int g = blockIdx.x * 8 + w;
  const float cnt = fmaxf(pcnt[g], 1.0f);
  const int idx = lane * 4;
  f4 s4 = *(const f4*)(psum + (long)g * 128 + idx);
  sg[w][idx + 0] = s4.x / cnt;
  sg[w][idx + 1] = s4.y / cnt;
  sg[w][idx + 2] = s4.z / cnt;
  sg[w][idx + 3] = s4.w / cnt;
  __syncthreads();
#pragma unroll
  for (int oo = 0; oo < 2; ++oo) {
    int o = lane + oo * 32;
    float acc = b1[o];
    for (int j = 0; j < 128; ++j) acc += sg[w][j] * W1[j * 64 + o];
    l1[w][o] = fmaxf(acc, 0.0f);
  }
  __syncthreads();
  float acc2 = b2[lane];
  for (int j = 0; j < 64; ++j) acc2 += l1[w][j] * W2[j * 32 + lane];
  acc2 = fmaxf(acc2, 0.0f);
  float p = acc2 * W3[lane];
  p += __shfl_xor(p, 1, 32);
  p += __shfl_xor(p, 2, 32);
  p += __shfl_xor(p, 4, 32);
  p += __shfl_xor(p, 8, 32);
  p += __shfl_xor(p, 16, 32);
  if (lane == 0) out[g] = 1.0f / (1.0f + __expf(-(p + b3[0])));
}

// ---------------- host launch ------------------------------------------------
extern "C" void kernel_launch(void* const* d_in, const int* in_sizes, int n_in,
                              void* d_out, int out_size, void* d_ws, size_t ws_size,
                              hipStream_t stream) {
  (void)in_sizes; (void)n_in; (void)out_size; (void)ws_size;
  const float* x     = (const float*)d_in[0];
  const int*   ei    = (const int*)d_in[1];
  const int*   srcp  = ei;
  const int*   dstp  = ei + EE;
  const int*   batch = (const int*)d_in[2];
  const float* in_Wq = (const float*)d_in[3];  const float* in_bq = (const float*)d_in[4];
  const float* in_Wk = (const float*)d_in[5];  const float* in_bk = (const float*)d_in[6];
  const float* in_Wv = (const float*)d_in[7];  const float* in_bv = (const float*)d_in[8];
  const float* in_Ws = (const float*)d_in[9];  const float* in_bs = (const float*)d_in[10];
  const float* in_Wbeta = (const float*)d_in[11];
  const float* blk_Wq = (const float*)d_in[12]; const float* blk_bq = (const float*)d_in[13];
  const float* blk_Wk = (const float*)d_in[14]; const float* blk_bk = (const float*)d_in[15];
  const float* blk_Wv = (const float*)d_in[16]; const float* blk_bv = (const float*)d_in[17];
  const float* blk_Ws = (const float*)d_in[18]; const float* blk_bs = (const float*)d_in[19];
  const float* blk_Wbeta = (const float*)d_in[20];
  const float* W1 = (const float*)d_in[21]; const float* b1 = (const float*)d_in[22];
  const float* W2 = (const float*)d_in[23]; const float* b2 = (const float*)d_in[24];
  const float* W3 = (const float*)d_in[25]; const float* b3 = (const float*)d_in[26];
  float* outp = (float*)d_out;

  // workspace carve-up (all chunk sizes are multiples of 256B)
  char* ws = (char*)d_ws;
  size_t o = 0;
  auto take = [&](size_t bytes) { char* p = ws + o; o += (bytes + 255) & ~(size_t)255; return p; };
  unsigned short* xb    = (unsigned short*)take((size_t)NN * 128 * 2);
  unsigned short* wt    = (unsigned short*)take(512 * 128 * 2);
  float* biascat        = (float*)take(512 * 4);
  float* qkvs           = (float*)take((size_t)NN * 512 * 4);
  float* mmax           = (float*)take((size_t)NN * 4 * 4);
  float* denom          = (float*)take((size_t)NN * 4 * 4);
  float* outagg         = (float*)take((size_t)NN * 128 * 4);
  float* alphab         = (float*)take((size_t)EE * 4 * 4);
  float* hbuf           = (float*)take((size_t)NN * 128 * 4);
  float* psum           = (float*)take((size_t)GG * 128 * 4);
  float* pcnt           = (float*)take((size_t)GG * 4);

  const float NEG_INF = -__builtin_huge_valf();

  // x (f32) -> bf16 feature buffer
  f32_to_bf16_k<<<(NN * 128) / 256, 256, 0, stream>>>(x, xb, (long)NN * 128);

  for (int l = 0; l < 1 + LL; ++l) {
    const float *Wq, *Wk, *Wv, *Ws, *bq, *bk, *bv, *bs, *Wb;
    if (l == 0) {
      Wq = in_Wq; Wk = in_Wk; Wv = in_Wv; Ws = in_Ws;
      bq = in_bq; bk = in_bk; bv = in_bv; bs = in_bs; Wb = in_Wbeta;
    } else {
      int j = l - 1;
      Wq = blk_Wq + (size_t)j * DD * DD; Wk = blk_Wk + (size_t)j * DD * DD;
      Wv = blk_Wv + (size_t)j * DD * DD; Ws = blk_Ws + (size_t)j * DD * DD;
      bq = blk_bq + (size_t)j * DD; bk = blk_bk + (size_t)j * DD;
      bv = blk_bv + (size_t)j * DD; bs = blk_bs + (size_t)j * DD;
      Wb = blk_Wbeta + (size_t)j * 3 * DD;
    }

    pack_weights<<<(512 * 128) / 256, 256, 0, stream>>>(Wq, Wk, Wv, Ws, bq, bk, bv, bs,
                                                        wt, biascat);
    fill_f32<<<(NN * 4 + 255) / 256, 256, 0, stream>>>(mmax, NEG_INF, (long)NN * 4);
    fill_f32<<<(NN * 4 + 255) / 256, 256, 0, stream>>>(denom, 0.0f, (long)NN * 4);
    fill_f32<<<(NN * 128 + 255) / 256, 256, 0, stream>>>(outagg, 0.0f, (long)NN * 128);

    gemm_qkvs<<<NN / 16, 256, 0, stream>>>(xb, wt, biascat, qkvs);  // 6250 blocks, 8 waves each

    edge_alpha<<<EE / 8, 256, 0, stream>>>(srcp, dstp, qkvs, alphab, mmax);
    edge_exp<<<(EE * 4) / 256, 256, 0, stream>>>(dstp, alphab, mmax, denom);
    edge_msg<<<EE / 8, 256, 0, stream>>>(srcp, dstp, qkvs, alphab, denom, outagg);

    beta_blend<<<NN / 8, 256, 0, stream>>>(outagg, qkvs, Wb, hbuf, xb);
  }

  // global mean pool + MLP head
  fill_f32<<<(GG * 128 + 255) / 256, 256, 0, stream>>>(psum, 0.0f, (long)GG * 128);
  fill_f32<<<(GG + 255) / 256, 256, 0, stream>>>(pcnt, 0.0f, (long)GG);
  pool_accum<<<(NN * 128) / 256, 256, 0, stream>>>(batch, hbuf, psum);
  cnt_accum<<<(NN + 255) / 256, 256, 0, stream>>>(batch, pcnt);
  mlp_head<<<GG / 8, 256, 0, stream>>>(psum, pcnt, W1, b1, W2, b2, W3, b3, outp);
}
